// BlockCFLChannel_24721831756344
// MI455X (gfx1250) — matrix operand order repaired and checked by reading, not compiled, not run on hardware
//
#include <hip/hip_runtime.h>
#include <math.h>

// Problem constants (from reference)
#define BB  8
#define CC  512
#define HH  16
#define WWW 16
#define LL  256            // H*W
#define DI  1024
#define NN  16
#define RR  32
#define KK  4
#define MR  (BB*LL)        // 2048 rows (b*L + l)

typedef __bf16 bf16;
typedef unsigned int u32;
typedef __attribute__((ext_vector_type(16))) __bf16 v16bf;
typedef __attribute__((ext_vector_type(8)))  float  v8f;
typedef __attribute__((ext_vector_type(4)))  u32    u32x4;

union Frag { v16bf v; u32x4 q[2]; };

__device__ __forceinline__ void load_frag_a(Frag& f, const bf16* row, int k0, int half) {
    f.q[0] = *reinterpret_cast<const u32x4*>(row + k0 + half * 8);
    f.q[1] = *reinterpret_cast<const u32x4*>(row + k0 + 16 + half * 8);
}
__device__ __forceinline__ void load_frag_b(Frag& f, const bf16* row, int k0, int half) {
    f.q[0] = *reinterpret_cast<const u32x4*>(row + k0 + half * 8);
    f.q[1] = *reinterpret_cast<const u32x4*>(row + k0 + 16 + half * 8);
}

// ---------------------------------------------------------------------------
// Block-wide 2-value reduction (256 threads)
// ---------------------------------------------------------------------------
__device__ __forceinline__ void block_reduce2(float& a, float& b) {
    __shared__ float sa[256];
    __shared__ float sb[256];
    sa[threadIdx.x] = a; sb[threadIdx.x] = b;
    __syncthreads();
    for (int s = 128; s > 0; s >>= 1) {
        if (threadIdx.x < (unsigned)s) {
            sa[threadIdx.x] += sa[threadIdx.x + s];
            sb[threadIdx.x] += sb[threadIdx.x + s];
        }
        __syncthreads();
    }
    a = sa[0]; b = sb[0];
    __syncthreads();
}

// ---------------------------------------------------------------------------
// WMMA GEMM: C[M,N] = A[M,K](bf16, row-major, lda) x W[N,K](bf16)^T
// Wave computes a 32x32 output (2x2 WMMA tiles, 4 accumulators) so each
// fragment load feeds two v_wmma ops. Block = 8 waves (4M x 2N) -> 128x64.
// M % 128 == 0, N % 64 == 0, K % 32 == 0 for all GEMMs in this model.
// ---------------------------------------------------------------------------
__global__ __launch_bounds__(256) void gemm_bf16_wmma(
    const bf16* __restrict__ A, int lda,
    const bf16* __restrict__ W, int Kd,
    float* __restrict__ C, int ldc,
    int M, int N)
{
    const int lane = threadIdx.x & 31;
    const int wave = threadIdx.x >> 5;
    const int wm = wave >> 1, wn = wave & 1;
    const int m0 = blockIdx.y * 128 + wm * 32;
    const int n0 = blockIdx.x * 64 + wn * 32;
    if (m0 >= M || n0 >= N) return;
    const int r    = lane & 15;
    const int half = lane >> 4;

    const bf16* arow0 = A + (size_t)(m0 + r) * lda;
    const bf16* arow1 = arow0 + (size_t)16 * lda;
    const bf16* wrow0 = W + (size_t)(n0 + r) * Kd;
    const bf16* wrow1 = wrow0 + (size_t)16 * Kd;

    v8f acc00 = {}, acc01 = {}, acc10 = {}, acc11 = {};
    for (int k0 = 0; k0 < Kd; k0 += 32) {
        __builtin_prefetch(arow0 + k0 + 128, 0, 1);
        __builtin_prefetch(arow1 + k0 + 128, 0, 1);
        Frag a0, a1, b0, b1;
        load_frag_a(a0, arow0, k0, half);
        load_frag_a(a1, arow1, k0, half);
        load_frag_b(b0, wrow0, k0, half);
        load_frag_b(b1, wrow1, k0, half);
        acc00 = __builtin_amdgcn_wmma_f32_16x16x32_bf16(
                    false, a0.v, false, b0.v, (short)0, acc00, false, false);
        acc01 = __builtin_amdgcn_wmma_f32_16x16x32_bf16(
                    false, a0.v, false, b1.v, (short)0, acc01, false, false);
        acc10 = __builtin_amdgcn_wmma_f32_16x16x32_bf16(
                    false, a1.v, false, b0.v, (short)0, acc10, false, false);
        acc11 = __builtin_amdgcn_wmma_f32_16x16x32_bf16(
                    false, a1.v, false, b1.v, (short)0, acc11, false, false);
    }
    float* c0 = C + (size_t)(m0 + half * 8) * ldc + n0 + r;
    float* c1 = C + (size_t)(m0 + 16 + half * 8) * ldc + n0 + r;
#pragma unroll
    for (int v = 0; v < 8; ++v) {
        c0[(size_t)v * ldc]      = acc00[v];
        c0[(size_t)v * ldc + 16] = acc01[v];
        c1[(size_t)v * ldc]      = acc10[v];
        c1[(size_t)v * ldc + 16] = acc11[v];
    }
}

// ---------------------------------------------------------------------------
// 3x3 conv (zero-pad) as 9 row-shifted WMMA GEMM taps, same 2x2 wave tiling.
// A: vss[b*256+p, C] bf16 ; W9: [9][O][I] bf16 ; out: [MR, C] f32
// ---------------------------------------------------------------------------
__global__ __launch_bounds__(256) void conv3x3_bf16_wmma(
    const bf16* __restrict__ A,
    const bf16* __restrict__ W9,
    float* __restrict__ C)
{
    const int lane = threadIdx.x & 31;
    const int wave = threadIdx.x >> 5;
    const int wm = wave >> 1, wn = wave & 1;
    const int m0 = blockIdx.y * 128 + wm * 32;
    const int n0 = blockIdx.x * 64 + wn * 32;
    const int r    = lane & 15;
    const int half = lane >> 4;

    const int mrow0 = m0 + r, mrow1 = m0 + 16 + r;
    const int b  = mrow0 >> 8;                  // 32-row tile stays in one batch
    const int p0 = mrow0 & 255, p1 = mrow1 & 255;
    const int h0 = p0 >> 4, w0 = p0 & 15;
    const int h1 = p1 >> 4, w1 = p1 & 15;

    v8f acc00 = {}, acc01 = {}, acc10 = {}, acc11 = {};
    for (int tap = 0; tap < 9; ++tap) {
        const int dy = tap / 3 - 1, dx = tap % 3 - 1;
        const int hh0 = h0 + dy, ww0 = w0 + dx;
        const int hh1 = h1 + dy, ww1 = w1 + dx;
        const bool v0 = (hh0 >= 0) & (hh0 < HH) & (ww0 >= 0) & (ww0 < WWW);
        const bool v1 = (hh1 >= 0) & (hh1 < HH) & (ww1 >= 0) & (ww1 < WWW);
        const bf16* arow0 = A + (size_t)(b * 256 + hh0 * 16 + ww0) * CC;
        const bf16* arow1 = A + (size_t)(b * 256 + hh1 * 16 + ww1) * CC;
        const bf16* wrow0 = W9 + (size_t)tap * CC * CC + (size_t)(n0 + r) * CC;
        const bf16* wrow1 = wrow0 + (size_t)16 * CC;
        for (int k0 = 0; k0 < CC; k0 += 32) {
            Frag a0, a1, b0, b1;
            if (v0) load_frag_a(a0, arow0, k0, half);
            else    { a0.q[0] = (u32x4)0; a0.q[1] = (u32x4)0; }
            if (v1) load_frag_a(a1, arow1, k0, half);
            else    { a1.q[0] = (u32x4)0; a1.q[1] = (u32x4)0; }
            load_frag_b(b0, wrow0, k0, half);
            load_frag_b(b1, wrow1, k0, half);
            acc00 = __builtin_amdgcn_wmma_f32_16x16x32_bf16(
                        false, a0.v, false, b0.v, (short)0, acc00, false, false);
            acc01 = __builtin_amdgcn_wmma_f32_16x16x32_bf16(
                        false, a0.v, false, b1.v, (short)0, acc01, false, false);
            acc10 = __builtin_amdgcn_wmma_f32_16x16x32_bf16(
                        false, a1.v, false, b0.v, (short)0, acc10, false, false);
            acc11 = __builtin_amdgcn_wmma_f32_16x16x32_bf16(
                        false, a1.v, false, b1.v, (short)0, acc11, false, false);
        }
    }
    float* c0 = C + (size_t)(m0 + half * 8) * CC + n0 + r;
    float* c1 = C + (size_t)(m0 + 16 + half * 8) * CC + n0 + r;
#pragma unroll
    for (int v = 0; v < 8; ++v) {
        c0[(size_t)v * CC]      = acc00[v];
        c0[(size_t)v * CC + 16] = acc01[v];
        c1[(size_t)v * CC]      = acc10[v];
        c1[(size_t)v * CC + 16] = acc11[v];
    }
}

// ---------------------------------------------------------------------------
// Elementwise / small kernels
// ---------------------------------------------------------------------------
__global__ void f32_to_bf16(const float* __restrict__ s, bf16* __restrict__ d, int n) {
    int i = blockIdx.x * 256 + threadIdx.x;
    if (i < n) d[i] = (bf16)s[i];
}

// (O,I,3,3) -> [tap][O][I] bf16
__global__ void reorder_conv9(const float* __restrict__ s, bf16* __restrict__ d) {
    int i = blockIdx.x * 256 + threadIdx.x;          // 9*512*512
    int ii  = i & 511;
    int o   = (i >> 9) & 511;
    int tap = i >> 18;
    d[i] = (bf16)s[((size_t)o * CC + ii) * 9 + tap];
}

// LN over C=512 on x transposed to (b,h,w,c); output bf16 [MR, C]
__global__ __launch_bounds__(256) void ln1_kernel(
    const float* __restrict__ x, const float* __restrict__ g,
    const float* __restrict__ b, bf16* __restrict__ xn)
{
    int m = blockIdx.x, bi = m >> 8, p = m & 255;
    const float* xb = x + (size_t)bi * CC * LL + p;
    int c0 = threadIdx.x, c1 = threadIdx.x + 256;
    float v0 = xb[(size_t)c0 * LL];
    float v1 = xb[(size_t)c1 * LL];
    float s = v0 + v1, s2 = v0 * v0 + v1 * v1;
    block_reduce2(s, s2);
    float mu  = s * (1.f / 512.f);
    float var = s2 * (1.f / 512.f) - mu * mu;
    float inv = rsqrtf(var + 1e-5f);
    xn[(size_t)m * CC + c0] = (bf16)((v0 - mu) * inv * g[c0] + b[c0]);
    xn[(size_t)m * CC + c1] = (bf16)((v1 - mu) * inv * g[c1] + b[c1]);
}

// depthwise 3x3 + bias + SiLU; xz[m, 0..Di) is xa; output xc_t bf16 [m, d]
__global__ void dwconv_silu(const float* __restrict__ xz,
                            const float* __restrict__ cw,
                            const float* __restrict__ cb,
                            bf16* __restrict__ xct)
{
    int i = blockIdx.x * 256 + threadIdx.x;          // MR*DI
    int d = i & (DI - 1);
    int m = i >> 10;
    int b = m >> 8, p = m & 255;
    int h = p >> 4, w = p & 15;
    float acc = cb[d];
#pragma unroll
    for (int ky = 0; ky < 3; ++ky)
#pragma unroll
        for (int kx = 0; kx < 3; ++kx) {
            int hh = h + ky - 1, ww = w + kx - 1;
            if (hh >= 0 && hh < HH && ww >= 0 && ww < WWW)
                acc += xz[(size_t)(b * 256 + hh * 16 + ww) * (2 * DI) + d] *
                       cw[d * 9 + ky * 3 + kx];
        }
    float s = acc / (1.f + __expf(-acc));
    xct[i] = (bf16)s;
}

// pull first R=32 cols of x_dbl into bf16 for the dt GEMM
__global__ void extract_dts(const float* __restrict__ xdbl, bf16* __restrict__ dts) {
    int i = blockIdx.x * 256 + threadIdx.x;          // KK*MR*RR = 262144
    int r = i & 31;
    int m = (i >> 5) & (MR - 1);
    int k = i >> 16;
    dts[i] = (bf16)xdbl[((size_t)k * MR + m) * 64 + r];
}

// delta = softplus(delta_pre + dt_b[k,d]) in place
__global__ void softplus_bias(float* __restrict__ delta, const float* __restrict__ dtb) {
    int i = blockIdx.x * 256 + threadIdx.x;          // KK*MR*DI = 8M
    int d = i & (DI - 1);
    int k = i >> 21;
    float x = delta[i] + dtb[k * DI + d];
    delta[i] = (x > 20.f) ? x : log1pf(__expf(x));
}

__device__ __forceinline__ int scan_perm(int k, int l) {
    // spatial position p read at step l for direction k
    int l2 = (k >= 2) ? (255 - l) : l;
    if (k & 1) return ((l2 & 15) << 4) | (l2 >> 4);  // wh transpose
    return l2;
}

// Selective scan: one lane per (b,k,d); 16 states in registers; L=256 steps.
__global__ __launch_bounds__(256) void ss2d_scan(
    const float* __restrict__ delta,   // [K][MR][DI]
    const bf16*  __restrict__ u_bf,    // xc_t [MR][DI]
    const float* __restrict__ xdbl,    // [K][MR][64] (cols 32..47 = B, 48..63 = C)
    const float* __restrict__ A_logs,  // [K*DI][N]
    const float* __restrict__ Ds,      // [K*DI]
    float* __restrict__ ys)            // [K][MR][DI]
{
    const int blk = blockIdx.x;                      // 128 = B*K*4
    const int d  = ((blk & 3) << 8) + threadIdx.x;
    const int k  = (blk >> 2) & 3;
    const int b  = blk >> 4;

    float Aa[NN];
#pragma unroll
    for (int n = 0; n < NN; ++n)
        Aa[n] = -__expf(A_logs[((size_t)(k * DI + d)) * NN + n]);
    const float Dv = Ds[k * DI + d];

    float hst[NN];
#pragma unroll
    for (int n = 0; n < NN; ++n) hst[n] = 0.f;

    __shared__ float sBC[2 * NN];
    const float* dbase = delta + ((size_t)k * MR + b * 256) * DI + d;
    const float* xd    = xdbl  + ((size_t)k * MR + b * 256) * 64;
    float*       ybase = ys    + ((size_t)k * MR + b * 256) * DI + d;

    for (int l = 0; l < LL; ++l) {
        if (threadIdx.x < 32) sBC[threadIdx.x] = xd[l * 64 + 32 + threadIdx.x];
        __syncthreads();
        float dl = dbase[(size_t)l * DI];
        int   p  = scan_perm(k, l);
        float ul = (float)u_bf[(size_t)(b * 256 + p) * DI + d];
        float du = dl * ul;
        float yv = 0.f;
#pragma unroll
        for (int n = 0; n < NN; ++n) {
            hst[n] = __expf(dl * Aa[n]) * hst[n] + du * sBC[n];
            yv += hst[n] * sBC[NN + n];
        }
        ybase[(size_t)l * DI] = yv + ul * Dv;
        __syncthreads();
    }
}

// combine 4 directions + out_norm LN + * SiLU(z) -> yc bf16 [MR, DI]
__global__ __launch_bounds__(256) void combine_ln_gate(
    const float* __restrict__ ys, const float* __restrict__ xz,
    const float* __restrict__ g,  const float* __restrict__ bb,
    bf16* __restrict__ yc)
{
    const int m = blockIdx.x, b = m >> 8, p = m & 255;
    const int q = ((p & 15) << 4) | (p >> 4);
    const size_t r0 = ((size_t)0 * MR + b * 256 + p)        * DI;
    const size_t r1 = ((size_t)1 * MR + b * 256 + q)        * DI;
    const size_t r2 = ((size_t)2 * MR + b * 256 + 255 - p)  * DI;
    const size_t r3 = ((size_t)3 * MR + b * 256 + 255 - q)  * DI;

    float v[4]; float s = 0.f, s2 = 0.f;
#pragma unroll
    for (int i = 0; i < 4; ++i) {
        int d = threadIdx.x + i * 256;
        float t = ys[r0 + d] + ys[r1 + d] + ys[r2 + d] + ys[r3 + d];
        v[i] = t; s += t; s2 += t * t;
    }
    block_reduce2(s, s2);
    float mu  = s * (1.f / 1024.f);
    float var = s2 * (1.f / 1024.f) - mu * mu;
    float inv = rsqrtf(var + 1e-5f);
#pragma unroll
    for (int i = 0; i < 4; ++i) {
        int d = threadIdx.x + i * 256;
        float zn = xz[(size_t)m * (2 * DI) + DI + d];
        float sz = zn / (1.f + __expf(-zn));
        float yv = (v[i] - mu) * inv * g[d] + bb[d];
        yc[(size_t)m * DI + d] = (bf16)(yv * sz);
    }
}

// vss = x1 + out_proj_result, stored bf16 [MR, C]
__global__ void add_x_convert(const float* __restrict__ vpre,
                              const float* __restrict__ x, bf16* __restrict__ vbf)
{
    int i = blockIdx.x * 256 + threadIdx.x;          // MR*CC
    int c = i & (CC - 1);
    int m = i >> 9;
    int b = m >> 8, p = m & 255;
    vbf[i] = (bf16)(vpre[i] + x[((size_t)b * CC + c) * LL + p]);
}

__global__ void bn_relu6(const float* __restrict__ h, const float* __restrict__ g,
                         const float* __restrict__ b, const float* __restrict__ mean,
                         const float* __restrict__ var, bf16* __restrict__ o)
{
    int i = blockIdx.x * 256 + threadIdx.x;
    int c = i & (CC - 1);
    float v = (h[i] - mean[c]) * rsqrtf(var[c] + 1e-5f) * g[c] + b[c];
    v = fminf(fmaxf(v, 0.f), 6.f);
    o[i] = (bf16)v;
}

__global__ void gelu_bf(const float* __restrict__ s, bf16* __restrict__ d) {
    int i = blockIdx.x * 256 + threadIdx.x;
    float x = s[i];
    d[i] = (bf16)(0.5f * x * (1.f + erff(x * 0.70710678118f)));
}

__global__ void pool_kernel(const float* __restrict__ y, float* __restrict__ pooled) {
    int i = blockIdx.x * 256 + threadIdx.x;          // BB*CC = 4096
    int c = i & (CC - 1), b = i >> 9;
    const float* yb = y + ((size_t)b * CC + c) * LL;
    float s = 0.f;
    for (int p = 0; p < LL; ++p) s += yb[p];
    pooled[i] = s * (1.f / (float)LL);
}

__global__ void lin1_kernel(const float* __restrict__ pooled,
                            const float* __restrict__ w, const float* __restrict__ b,
                            float* __restrict__ hid)
{
    int j = threadIdx.x;                             // 256
    int bi = blockIdx.x;                             // 8
    const float* pr = pooled + (size_t)bi * CC;
    const float* wr = w + (size_t)j * CC;
    float s = b[j];
    for (int c = 0; c < CC; ++c) s += pr[c] * wr[c];
    hid[bi * 256 + j] = fmaxf(s, 0.f);
}

__global__ void lin2_gate_kernel(const float* __restrict__ hid,
                                 const float* __restrict__ w, const float* __restrict__ b,
                                 float* __restrict__ gate)
{
    int i = blockIdx.x * 256 + threadIdx.x;          // BB*CC
    int c = i & (CC - 1), bi = i >> 9;
    const float* hr = hid + (size_t)bi * 256;
    const float* wr = w + (size_t)c * 256;
    float s = b[c];
    for (int j = 0; j < 256; ++j) s += hr[j] * wr[j];
    gate[i] = 1.f / (1.f + __expf(-2.f * s));
}

__global__ void final_kernel(const float* __restrict__ xm,
                             const float* __restrict__ gate, float* __restrict__ out)
{
    int i = blockIdx.x * 256 + threadIdx.x;          // BB*CC*LL = 1M
    int p = i & 255;
    int c = (i >> 8) & (CC - 1);
    int b = i >> 17;
    out[i] = gate[b * CC + c] * xm[(size_t)(b * 256 + p) * CC + c];
}

// ---------------------------------------------------------------------------
// Host launcher
// ---------------------------------------------------------------------------
extern "C" void kernel_launch(void* const* d_in, const int* in_sizes, int n_in,
                              void* d_out, int out_size, void* d_ws, size_t ws_size,
                              hipStream_t stream) {
    (void)in_sizes; (void)n_in; (void)out_size; (void)ws_size;
    const float* x      = (const float*)d_in[0];
    const float* yin    = (const float*)d_in[1];
    const float* ln1_g  = (const float*)d_in[2];
    const float* ln1_b  = (const float*)d_in[3];
    const float* in_w   = (const float*)d_in[4];
    const float* conv_w = (const float*)d_in[5];
    const float* conv_b = (const float*)d_in[6];
    const float* xp_w   = (const float*)d_in[7];
    const float* dt_w   = (const float*)d_in[8];
    const float* dt_b   = (const float*)d_in[9];
    const float* A_logs = (const float*)d_in[10];
    const float* Ds     = (const float*)d_in[11];
    const float* on_g   = (const float*)d_in[12];
    const float* on_b   = (const float*)d_in[13];
    const float* op_w   = (const float*)d_in[14];
    const float* fcw    = (const float*)d_in[15];
    const float* bn_g   = (const float*)d_in[16];
    const float* bn_b   = (const float*)d_in[17];
    const float* bn_m   = (const float*)d_in[18];
    const float* bn_v   = (const float*)d_in[19];
    const float* fc1    = (const float*)d_in[20];
    const float* fc2    = (const float*)d_in[21];
    const float* l1w    = (const float*)d_in[22];
    const float* l1b    = (const float*)d_in[23];
    const float* l2w    = (const float*)d_in[24];
    const float* l2b    = (const float*)d_in[25];
    float* out = (float*)d_out;

    char* ws = (char*)d_ws;
    size_t off = 0;
    auto take = [&](size_t bytes) -> void* {
        void* p = ws + off;
        off = (off + bytes + 255) & ~(size_t)255;
        return p;
    };

    bf16* w_in   = (bf16*)take((size_t)2 * DI * CC * 2);
    bf16* w_xp   = (bf16*)take((size_t)KK * 64 * DI * 2);
    bf16* w_dt   = (bf16*)take((size_t)KK * DI * RR * 2);
    bf16* w_op   = (bf16*)take((size_t)CC * DI * 2);
    bf16* w_c9   = (bf16*)take((size_t)9 * CC * CC * 2);
    bf16* w_f1   = (bf16*)take((size_t)CC * CC * 2);
    bf16* w_f2   = (bf16*)take((size_t)CC * CC * 2);
    bf16* xn     = (bf16*)take((size_t)MR * CC * 2);
    float* xz    = (float*)take((size_t)MR * 2 * DI * 4);
    bf16* xct    = (bf16*)take((size_t)MR * DI * 2);
    float* xdbl  = (float*)take((size_t)KK * MR * 64 * 4);
    bf16* dtsbf  = (bf16*)take((size_t)KK * MR * RR * 2);
    float* delta = (float*)take((size_t)KK * MR * DI * 4);
    float* ysbuf = (float*)take((size_t)KK * MR * DI * 4);
    bf16* ycbf   = (bf16*)take((size_t)MR * DI * 2);
    float* vpre  = (float*)take((size_t)MR * CC * 4);
    bf16* vssbf  = (bf16*)take((size_t)MR * CC * 2);
    float* hconv = (float*)take((size_t)MR * CC * 4);
    bf16* hbnbf  = (bf16*)take((size_t)MR * CC * 2);
    float* f1o   = (float*)take((size_t)MR * CC * 4);
    bf16* g1bf   = (bf16*)take((size_t)MR * CC * 2);
    float* xm    = (float*)take((size_t)MR * CC * 4);
    float* pooled= (float*)take((size_t)BB * CC * 4);
    float* hid   = (float*)take((size_t)BB * 256 * 4);
    float* gate  = (float*)take((size_t)BB * CC * 4);

    auto blocks = [](int n) { return (n + 255) / 256; };

    // --- weight conversion (f32 -> bf16) ---
    f32_to_bf16<<<blocks(2 * DI * CC), 256, 0, stream>>>(in_w, w_in, 2 * DI * CC);
    f32_to_bf16<<<blocks(KK * 64 * DI), 256, 0, stream>>>(xp_w, w_xp, KK * 64 * DI);
    f32_to_bf16<<<blocks(KK * DI * RR), 256, 0, stream>>>(dt_w, w_dt, KK * DI * RR);
    f32_to_bf16<<<blocks(CC * DI), 256, 0, stream>>>(op_w, w_op, CC * DI);
    f32_to_bf16<<<blocks(CC * CC), 256, 0, stream>>>(fc1, w_f1, CC * CC);
    f32_to_bf16<<<blocks(CC * CC), 256, 0, stream>>>(fc2, w_f2, CC * CC);
    reorder_conv9<<<blocks(9 * CC * CC), 256, 0, stream>>>(fcw, w_c9);

    // --- LN1 + in_proj GEMM (2048x2048x512) ---
    ln1_kernel<<<MR, 256, 0, stream>>>(x, ln1_g, ln1_b, xn);
    gemm_bf16_wmma<<<dim3(2 * DI / 64, MR / 128), 256, 0, stream>>>(
        xn, CC, w_in, CC, xz, 2 * DI, MR, 2 * DI);

    // --- depthwise conv + SiLU -> xc_t (channels-last, bf16) ---
    dwconv_silu<<<blocks(MR * DI), 256, 0, stream>>>(xz, conv_w, conv_b, xct);

    // --- x_proj per direction (2048x64x1024), then dt GEMM (2048x1024x32) ---
    for (int k = 0; k < KK; ++k) {
        gemm_bf16_wmma<<<dim3(1, MR / 128), 256, 0, stream>>>(
            xct, DI, w_xp + (size_t)k * 64 * DI, DI,
            xdbl + (size_t)k * MR * 64, 64, MR, 64);
    }
    extract_dts<<<blocks(KK * MR * RR), 256, 0, stream>>>(xdbl, dtsbf);
    for (int k = 0; k < KK; ++k) {
        gemm_bf16_wmma<<<dim3(DI / 64, MR / 128), 256, 0, stream>>>(
            dtsbf + (size_t)k * MR * RR, RR, w_dt + (size_t)k * DI * RR, RR,
            delta + (size_t)k * MR * DI, DI, MR, DI);
    }
    softplus_bias<<<blocks(KK * MR * DI), 256, 0, stream>>>(delta, dt_b);

    // --- selective scan (parallel over 32K channels, serial L=256) ---
    ss2d_scan<<<BB * KK * (DI / 256), 256, 0, stream>>>(
        delta, xct, xdbl, A_logs, Ds, ysbuf);

    // --- combine directions + out_norm + SiLU(z) gate ---
    combine_ln_gate<<<MR, 256, 0, stream>>>(ysbuf, xz, on_g, on_b, ycbf);

    // --- out_proj GEMM (2048x512x1024) + residual ---
    gemm_bf16_wmma<<<dim3(CC / 64, MR / 128), 256, 0, stream>>>(
        ycbf, DI, w_op, DI, vpre, CC, MR, CC);
    add_x_convert<<<blocks(MR * CC), 256, 0, stream>>>(vpre, x, vssbf);

    // --- conv-FFN: 3x3 conv (WMMA), BN+ReLU6, fc1+GELU, fc2 ---
    conv3x3_bf16_wmma<<<dim3(CC / 64, MR / 128), 256, 0, stream>>>(vssbf, w_c9, hconv);
    bn_relu6<<<blocks(MR * CC), 256, 0, stream>>>(hconv, bn_g, bn_b, bn_m, bn_v, hbnbf);
    gemm_bf16_wmma<<<dim3(CC / 64, MR / 128), 256, 0, stream>>>(
        hbnbf, CC, w_f1, CC, f1o, CC, MR, CC);
    gelu_bf<<<blocks(MR * CC), 256, 0, stream>>>(f1o, g1bf);
    gemm_bf16_wmma<<<dim3(CC / 64, MR / 128), 256, 0, stream>>>(
        g1bf, CC, w_f2, CC, xm, CC, MR, CC);

    // --- SE-style gate from pooled y ---
    pool_kernel<<<blocks(BB * CC), 256, 0, stream>>>(yin, pooled);
    lin1_kernel<<<BB, 256, 0, stream>>>(pooled, l1w, l1b, hid);
    lin2_gate_kernel<<<blocks(BB * CC), 256, 0, stream>>>(hid, l2w, l2b, gate);

    // --- final output [B,C,H,W] ---
    final_kernel<<<blocks(BB * CC * LL), 256, 0, stream>>>(xm, gate, out);
}
